// NonLocalBlock_30666066493500
// MI455X (gfx1250) — matrix operand order repaired and checked by reading, not compiled, not run on hardware
//
#include <hip/hip_runtime.h>

#define Bc 8
#define Cdim 256
#define Kdim 128
#define Ndim 4096
#define WSZ 32768   // elements per weight matrix (128*256)

typedef __attribute__((ext_vector_type(16))) __bf16 v16bf;
typedef __attribute__((ext_vector_type(8)))  __bf16 v8bf;
typedef __attribute__((ext_vector_type(8)))  float  v8f;
typedef __attribute__((ext_vector_type(4)))  unsigned int u32x4;
typedef __attribute__((ext_vector_type(8)))  unsigned int u32x8;

union FragA { v16bf v; v8bf h[2]; };

__device__ __forceinline__ v8f wmma_bf16(v16bf a, v16bf b, v8f c) {
  return __builtin_amdgcn_wmma_f32_16x16x32_bf16(false, a, false, b, (short)0, c,
                                                 false, false);
}

__device__ __forceinline__ void wait_tensorcnt0() {
#if __has_builtin(__builtin_amdgcn_s_wait_tensorcnt)
  __builtin_amdgcn_s_wait_tensorcnt(0);
#else
  asm volatile("s_wait_tensorcnt 0x0" ::: "memory");
#endif
}

// Issue one 2D TDM tile load global->LDS (D# groups 0+1, VADDR2/3 = NULL).
// Dims/strides in data_size=8B units; LDS padding per D# pad fields.
__device__ __forceinline__ void tdm_load_2d(unsigned lds_addr, const void* gaddr,
                                            unsigned tile_d0, unsigned tile_d1,
                                            unsigned long long stride_d0,
                                            unsigned pad_interval_code,
                                            unsigned pad_amount_code) {
  const unsigned long long ga = (unsigned long long)(size_t)gaddr;
  u32x4 g0;
  g0[0] = 1u;                                          // count=1, user descriptor
  g0[1] = lds_addr;                                    // LDS byte address
  g0[2] = (unsigned)(ga & 0xFFFFFFFFu);                // global_addr[31:0]
  g0[3] = (unsigned)((ga >> 32) & 0x1FFFFFFu) | (2u << 30);  // addr[56:32] | type=2
  u32x8 g1;
  g1[0] = (3u << 16) | (1u << 20) |                    // data_size=8B, pad_enable
          (pad_interval_code << 22) | (pad_amount_code << 25);
  g1[1] = (tile_d0 & 0xFFFFu) << 16;                   // tensor_dim0[15:0]
  g1[2] = ((tile_d0 >> 16) & 0xFFFFu) |                // tensor_dim0[31:16]
          ((tile_d1 & 0xFFFFu) << 16);                 // tensor_dim1[15:0]
  g1[3] = ((tile_d1 >> 16) & 0xFFFFu) |                // tensor_dim1[31:16]
          ((tile_d0 & 0xFFFFu) << 16);                 // tile_dim0
  g1[4] = (tile_d1 & 0xFFFFu);                         // tile_dim1 (tile_dim2=0)
  g1[5] = (unsigned)(stride_d0 & 0xFFFFFFFFu);         // tensor_dim0_stride lo
  g1[6] = (unsigned)((stride_d0 >> 32) & 0xFFFFu);     // stride hi16 (dim1_stride=0)
  g1[7] = 0u;
  asm volatile("tensor_load_to_lds %0, %1" :: "s"(g0), "s"(g1) : "memory");
}

// -------------------------------------------------------------------------
// Kernel W: one-time f32 -> bf16 conversion of all four weight matrices.
// -------------------------------------------------------------------------
__global__ __launch_bounds__(256)
void cvtw_kernel(const float* __restrict__ tw, const float* __restrict__ pw,
                 const float* __restrict__ gw, const float* __restrict__ ow,
                 __bf16* __restrict__ wbf) {
  const int i = blockIdx.x * 256 + threadIdx.x;        // 0..32767
  wbf[i]           = (__bf16)tw[i];
  wbf[WSZ + i]     = (__bf16)pw[i];
  wbf[2 * WSZ + i] = (__bf16)gw[i];
  wbf[3 * WSZ + i] = (__bf16)ow[i];
}

// -------------------------------------------------------------------------
// Kernel A: 1x1-conv projections -> thetaN/phiN [B][N][128], gT [B][128][N] (bf16)
// -------------------------------------------------------------------------
__global__ __launch_bounds__(256)
void proj_kernel(const float* __restrict__ x, const __bf16* __restrict__ wbf,
                 const float* __restrict__ tb, const float* __restrict__ pb,
                 const float* __restrict__ gb,
                 __bf16* __restrict__ thetaN, __bf16* __restrict__ phiN,
                 __bf16* __restrict__ gT) {
  __shared__ __align__(16) __bf16 sX[64][Cdim + 8];    // x^T tile, padded
  const int tid = threadIdx.x;
  const int b  = blockIdx.x >> 6;
  const int n0 = (blockIdx.x & 63) * 64;

  for (int it = 0; it < 64; ++it) {
    const int idx = tid + it * 256;
    const int c = idx >> 6, n = idx & 63;
    sX[n][c] = (__bf16)x[((size_t)b * Cdim + c) * Ndim + n0 + n];
  }
  __syncthreads();

  const int wv = tid >> 5, lane = tid & 31;
  const int nt = wv & 3, kh = wv >> 2;
  const int lrow = lane & 15, lhi = lane >> 4;

  FragA a[8];
#pragma unroll
  for (int cc = 0; cc < 8; ++cc) {
    const __bf16* p = &sX[nt * 16 + lrow][cc * 32 + lhi * 16];
    a[cc].h[0] = *(const v8bf*)p;
    a[cc].h[1] = *(const v8bf*)(p + 8);
  }

  const float* Bv[3] = {tb, pb, gb};
#pragma unroll
  for (int pj = 0; pj < 3; ++pj) {
#pragma unroll
    for (int kt = 0; kt < 4; ++kt) {
      const int k0 = (kh * 4 + kt) * 16;
      const int kcol = k0 + lrow;
      v8f acc = {};
#pragma unroll
      for (int cc = 0; cc < 8; ++cc) {
        // B frag: fixed k, 16 contiguous c values, bf16 straight from global
        const __bf16* wp = wbf + (size_t)pj * WSZ + (size_t)kcol * Cdim
                           + cc * 32 + lhi * 16;
        FragA bf;
        bf.h[0] = *(const v8bf*)wp;
        bf.h[1] = *(const v8bf*)(wp + 8);
        acc = wmma_bf16(a[cc].v, bf.v, acc);
      }
      const float bias = Bv[pj][kcol];
#pragma unroll
      for (int j = 0; j < 8; ++j) {
        const int n = n0 + nt * 16 + lhi * 8 + j;
        const __bf16 val = (__bf16)(acc[j] + bias);
        if (pj == 0)      thetaN[((size_t)b * Ndim + n) * Kdim + kcol] = val;
        else if (pj == 1) phiN  [((size_t)b * Ndim + n) * Kdim + kcol] = val;
        else              gT    [((size_t)b * Kdim + kcol) * Ndim + n] = val;
      }
    }
  }
}

// -------------------------------------------------------------------------
// Kernel B: flash attention. 8 waves / 128 query rows per block; keys
// streamed in 32-wide tiles via double-buffered TDM (tensor_load_to_lds).
// -------------------------------------------------------------------------
__global__ __launch_bounds__(256)
void attn_kernel(const __bf16* __restrict__ thetaN, const __bf16* __restrict__ phiN,
                 const __bf16* __restrict__ gT, __bf16* __restrict__ yN) {
  __shared__ __align__(16) __bf16 sK[2][32][Kdim + 8]; // keys x dims (+16B pad/row)
  __shared__ __align__(16) __bf16 sG[2][Kdim][40];     // dims x keys (+16B pad/row)
  __shared__ __align__(16) __bf16 sP[8][16][40];       // per-wave P round-trip
  const int tid = threadIdx.x;
  const int b  = blockIdx.x >> 5;
  const int n0 = (blockIdx.x & 31) * 128;
  const int wv = tid >> 5, lane = tid & 31;
  const int lrow = lane & 15, lhi = lane >> 4;
  const int qrow = n0 + wv * 16;
  const bool leader = ((__builtin_amdgcn_readfirstlane(tid) >> 5) == 0);

  // Persistent Q fragments (4 chunks of head-dim 32).
  FragA qf[4];
#pragma unroll
  for (int cc = 0; cc < 4; ++cc) {
    const __bf16* p = thetaN + ((size_t)b * Ndim + qrow + lrow) * Kdim
                      + cc * 32 + lhi * 16;
    qf[cc].h[0] = *(const v8bf*)p;
    qf[cc].h[1] = *(const v8bf*)(p + 8);
  }

  float m[8], l[8];
  v8f y[8];
  const v8f vzero = {};
#pragma unroll
  for (int j = 0; j < 8; ++j) { m[j] = -3.0e38f; l[j] = 0.f; }
#pragma unroll
  for (int o = 0; o < 8; ++o) y[o] = vzero;

  auto issue_tile = [&](int buf, int kb) {
    const int kbase = kb * 32;
    // phi tile: 32 rows x 256B, contiguous rows; pad 16B per 256B (64DW->code5, 4DW->3)
    tdm_load_2d((unsigned)(size_t)&sK[buf][0][0],
                phiN + ((size_t)b * Ndim + kbase) * Kdim,
                32u, 32u, 32ull, 5u, 3u);
    // g tile: 128 rows x 64B, row stride 8192B; pad 16B per 64B (16DW->code3, 4DW->3)
    tdm_load_2d((unsigned)(size_t)&sG[buf][0][0],
                gT + (size_t)b * Kdim * Ndim + kbase,
                8u, 128u, 1024ull, 3u, 3u);
  };

  const int NB = Ndim / 32;
  if (leader) issue_tile(0, 0);

  for (int kb = 0; kb < NB; ++kb) {
    const int cur = kb & 1;
    if (leader) wait_tensorcnt0();
    __syncthreads();                              // tile `cur` ready; `cur^1` free
    if (leader && (kb + 1 < NB)) issue_tile(cur ^ 1, kb + 1);

    // S = Q * K^T (two 16-key D tiles).
    v8f S0 = {}, S1 = {};
#pragma unroll
    for (int cc = 0; cc < 4; ++cc) {
      FragA kf;
      const __bf16* p0 = &sK[cur][lrow][cc * 32 + lhi * 16];
      kf.h[0] = *(const v8bf*)p0; kf.h[1] = *(const v8bf*)(p0 + 8);
      S0 = wmma_bf16(qf[cc].v, kf.v, S0);
      const __bf16* p1 = &sK[cur][16 + lrow][cc * 32 + lhi * 16];
      kf.h[0] = *(const v8bf*)p1; kf.h[1] = *(const v8bf*)(p1 + 8);
      S1 = wmma_bf16(qf[cc].v, kf.v, S1);
    }

    // Online softmax (lane holds rows M=lhi*8+j; reduce across 16 lanes).
    float scale[8];
#pragma unroll
    for (int j = 0; j < 8; ++j) {
      float v = fmaxf(S0[j], S1[j]);
#pragma unroll
      for (int msk = 1; msk < 16; msk <<= 1) v = fmaxf(v, __shfl_xor(v, msk, 32));
      const float mn = fmaxf(m[j], v);
      scale[j] = __expf(m[j] - mn);
      const float p0 = __expf(S0[j] - mn);
      const float p1 = __expf(S1[j] - mn);
      S0[j] = p0; S1[j] = p1;
      float rs = p0 + p1;
#pragma unroll
      for (int msk = 1; msk < 16; msk <<= 1) rs += __shfl_xor(rs, msk, 32);
      l[j] = l[j] * scale[j] + rs;
      m[j] = mn;
    }
#pragma unroll
    for (int o = 0; o < 8; ++o)
#pragma unroll
      for (int j = 0; j < 8; ++j) y[o][j] *= scale[j];

    // P (C/D layout) -> LDS -> A fragment.
#pragma unroll
    for (int j = 0; j < 8; ++j) {
      sP[wv][lhi * 8 + j][lrow]      = (__bf16)S0[j];
      sP[wv][lhi * 8 + j][16 + lrow] = (__bf16)S1[j];
    }
    FragA pf;
    {
      const __bf16* p = &sP[wv][lrow][lhi * 16];
      pf.h[0] = *(const v8bf*)p;
      pf.h[1] = *(const v8bf*)(p + 8);
    }
    // y += P * g
#pragma unroll
    for (int o = 0; o < 8; ++o) {
      FragA gf;
      const __bf16* p = &sG[cur][o * 16 + lrow][lhi * 16];
      gf.h[0] = *(const v8bf*)p; gf.h[1] = *(const v8bf*)(p + 8);
      y[o] = wmma_bf16(pf.v, gf.v, y[o]);
    }
    __syncthreads();                              // done reading tile `cur`
  }

  // Epilogue: normalize, store y bf16 [B][N][128].
  float inv[8];
#pragma unroll
  for (int j = 0; j < 8; ++j) inv[j] = 1.0f / l[j];
#pragma unroll
  for (int o = 0; o < 8; ++o)
#pragma unroll
    for (int j = 0; j < 8; ++j) {
      const int n = qrow + lhi * 8 + j;
      yN[((size_t)b * Ndim + n) * Kdim + o * 16 + lrow] = (__bf16)(y[o][j] * inv[j]);
    }
}

// -------------------------------------------------------------------------
// Kernel C: out = out_w @ y^T + out_b + x (f32 output, fused residual).
// -------------------------------------------------------------------------
__global__ __launch_bounds__(256)
void out_kernel(const float* __restrict__ x, const __bf16* __restrict__ wbf,
                const float* __restrict__ ob, const __bf16* __restrict__ yN,
                float* __restrict__ out) {
  const int tid = threadIdx.x;
  const int b  = blockIdx.x >> 6;
  const int n0 = (blockIdx.x & 63) * 64;
  const int wv = tid >> 5, lane = tid & 31;
  const int lrow = lane & 15, lhi = lane >> 4;
  const int cbase = wv * 32;
  const __bf16* owb = wbf + 3 * WSZ;               // out_w bf16 [256][128]

  FragA a[8];
#pragma unroll
  for (int ct = 0; ct < 2; ++ct)
#pragma unroll
    for (int cc = 0; cc < 4; ++cc) {
      const __bf16* wp = owb + (size_t)(cbase + ct * 16 + lrow) * Kdim
                         + cc * 32 + lhi * 16;
      FragA f;
      f.h[0] = *(const v8bf*)wp;
      f.h[1] = *(const v8bf*)(wp + 8);
      a[ct * 4 + cc] = f;
    }

#pragma unroll
  for (int nt = 0; nt < 4; ++nt) {
    FragA bf[4];
#pragma unroll
    for (int cc = 0; cc < 4; ++cc) {
      const __bf16* p = yN + ((size_t)b * Ndim + n0 + nt * 16 + lrow) * Kdim
                        + cc * 32 + lhi * 16;
      bf[cc].h[0] = *(const v8bf*)p;
      bf[cc].h[1] = *(const v8bf*)(p + 8);
    }
#pragma unroll
    for (int ct = 0; ct < 2; ++ct) {
      v8f acc = {};
#pragma unroll
      for (int cc = 0; cc < 4; ++cc)
        acc = wmma_bf16(a[ct * 4 + cc].v, bf[cc].v, acc);
#pragma unroll
      for (int j = 0; j < 8; ++j) {
        const int c = cbase + ct * 16 + lhi * 8 + j;
        const int n = n0 + nt * 16 + lrow;
        const size_t idx = ((size_t)b * Cdim + c) * Ndim + n;
        out[idx] = acc[j] + ob[c] + x[idx];
      }
    }
  }
}

// -------------------------------------------------------------------------
extern "C" void kernel_launch(void* const* d_in, const int* in_sizes, int n_in,
                              void* d_out, int out_size, void* d_ws, size_t ws_size,
                              hipStream_t stream) {
  const float* x  = (const float*)d_in[0];
  const float* tw = (const float*)d_in[1];
  const float* tb = (const float*)d_in[2];
  const float* pw = (const float*)d_in[3];
  const float* pb = (const float*)d_in[4];
  const float* gw = (const float*)d_in[5];
  const float* gb = (const float*)d_in[6];
  const float* ow = (const float*)d_in[7];
  const float* ob = (const float*)d_in[8];
  float* out = (float*)d_out;

  const size_t SZ = (size_t)Bc * Ndim * Kdim;
  __bf16* thetaN = (__bf16*)d_ws;
  __bf16* phiN   = thetaN + SZ;
  __bf16* gT     = phiN + SZ;
  __bf16* yN     = gT + SZ;
  __bf16* wbf    = yN + SZ;                       // 4 * 32768 bf16 weights

  cvtw_kernel<<<dim3(WSZ / 256), dim3(256), 0, stream>>>(tw, pw, gw, ow, wbf);
  proj_kernel<<<dim3(Bc * (Ndim / 64)), dim3(256), 0, stream>>>(
      x, wbf, tb, pb, gb, thetaN, phiN, gT);
  attn_kernel<<<dim3(Bc * (Ndim / 128)), dim3(256), 0, stream>>>(
      thetaN, phiN, gT, yN);
  out_kernel<<<dim3(Bc * (Ndim / 64)), dim3(256), 0, stream>>>(
      x, wbf, ob, yN, out);
}
